// Qwen3VLMoeTextAttention_37812892074073
// MI455X (gfx1250) — matrix operand order repaired and checked by reading, not compiled, not run on hardware
//
#include <hip/hip_runtime.h>
#include <hip/hip_bf16.h>

// ---------------- problem constants ----------------
#define TT   2048
#define DD   2048
#define NQ   32
#define NKV  4
#define HH   128
#define NHD  (NQ * HH)    // 4096
#define KHD  (NKV * HH)   // 512

typedef __attribute__((ext_vector_type(16))) __bf16 v16bf;
typedef __attribute__((ext_vector_type(8)))  __bf16 v8bf;
typedef __attribute__((ext_vector_type(8)))  float  v8f;

// round-to-nearest-even f32 -> bf16
__device__ inline unsigned short f32_to_bf16u(float f) {
    unsigned int u = __builtin_bit_cast(unsigned int, f);
    unsigned int r = u + 0x7FFFu + ((u >> 16) & 1u);
    return (unsigned short)(r >> 16);
}
__device__ inline __bf16 f32_to_bf16(float f) {
    unsigned short s = f32_to_bf16u(f);
    return __builtin_bit_cast(__bf16, s);
}

// A-operand fragment (16-bit, 16x32): per lane one contiguous 16-element run
// at (base + half*16).
__device__ inline v16bf load_a_frag(const __bf16* p) {
    return *(const v16bf*)(p);
}

// ---------------- f32 -> bf16 flat convert ----------------
__global__ void cvt_bf16_kernel(const float* __restrict__ s,
                                unsigned short* __restrict__ d, int n) {
    int i = blockIdx.x * 256 + threadIdx.x;
    if (i < n) d[i] = f32_to_bf16u(s[i]);
}

// ---------------- f32 -> bf16 convert + transpose ----------------
// s is [rows][cols] f32 ; d is [cols][rows] bf16.  Writes coalesced.
__global__ void cvt_bf16_t_kernel(const float* __restrict__ s,
                                  unsigned short* __restrict__ d,
                                  int rows, int cols) {
    int idx = blockIdx.x * 256 + threadIdx.x;
    if (idx >= rows * cols) return;
    int c = idx / rows;          // output row (former column)
    int r = idx - c * rows;      // output col (former row)
    d[idx] = f32_to_bf16u(s[(size_t)r * cols + c]);
}

// ---------------- bf16 WMMA GEMM, B pre-transposed ----------------
// C[M x Ncols] (f32) = A[M x Kdim] (bf16 row-major) * BT[Ncols x Kdim]^T
// grid = (M/32, Ncols/256), block = 128 (4 waves).
// Each wave computes a 32x64 tile: 2 M-tiles x 4 N-tiles, 8 WMMAs per K-step.
__global__ __launch_bounds__(128) void gemm_bf16_tn(
    const unsigned short* __restrict__ A,
    const unsigned short* __restrict__ BT,
    float* __restrict__ C, int Kdim, int Ncols)
{
    const int tid  = threadIdx.x;
    const int wave = tid >> 5;
    const int lane = tid & 31;
    const int hl   = lane >> 4;
    const int l16  = lane & 15;
    const int m0   = blockIdx.x * 32;
    const int nw   = blockIdx.y * 256 + wave * 64;

    v8f acc[2][4];
#pragma unroll
    for (int mt = 0; mt < 2; ++mt)
#pragma unroll
        for (int nt = 0; nt < 4; ++nt) { v8f z = {}; acc[mt][nt] = z; }

    const __bf16* Ab  = (const __bf16*)A;
    const __bf16* Bb  = (const __bf16*)BT;

    for (int k0 = 0; k0 < Kdim; k0 += 32) {
        v16bf a[2];
#pragma unroll
        for (int mt = 0; mt < 2; ++mt)
            a[mt] = load_a_frag(Ab + (size_t)(m0 + mt * 16 + l16) * Kdim + k0 + hl * 16);

#pragma unroll
        for (int nt = 0; nt < 4; ++nt) {
            // B fragment: col = nw+nt*16+l16, contract kk = hl*16 + j (contiguous)
            v16bf b = *(const v16bf*)(Bb + (size_t)(nw + nt * 16 + l16) * Kdim
                                          + k0 + hl * 16);
#pragma unroll
            for (int mt = 0; mt < 2; ++mt)
                acc[mt][nt] = __builtin_amdgcn_wmma_f32_16x16x32_bf16(
                    false, a[mt], false, b, (short)0, acc[mt][nt], false, false);
        }
    }

#pragma unroll
    for (int mt = 0; mt < 2; ++mt)
#pragma unroll
        for (int nt = 0; nt < 4; ++nt) {
            const int ccol = nw + nt * 16 + l16;
#pragma unroll
            for (int r = 0; r < 8; ++r)
                C[(size_t)(m0 + mt * 16 + r + hl * 8) * Ncols + ccol] = acc[mt][nt][r];
        }
}

// ---------------- RMSNorm + RoPE, write bf16 [head][T][H] ----------------
// grid = (T, heads), block = 32 (one wave per (t, head) vector of 128)
__global__ __launch_bounds__(32) void qk_post_kernel(
    const float* __restrict__ x,      // [T][heads*128]
    const float* __restrict__ cosT,   // [T][128]
    const float* __restrict__ sinT,   // [T][128]
    const float* __restrict__ w,      // [128]
    unsigned short* __restrict__ out, // [heads][T][128]
    int heads)
{
    const int t    = blockIdx.x;
    const int hd   = blockIdx.y;
    const int lane = threadIdx.x;
    const float* xp = x + ((size_t)t * heads + hd) * HH;

    float v[4], ss = 0.0f;
#pragma unroll
    for (int i = 0; i < 4; ++i) { v[i] = xp[lane + i * 32]; ss += v[i] * v[i]; }
#pragma unroll
    for (int m = 16; m >= 1; m >>= 1) ss += __shfl_xor(ss, m, 32);
    const float inv = rsqrtf(ss * (1.0f / HH) + 1e-6f);

    float y[4];
#pragma unroll
    for (int i = 0; i < 4; ++i) y[i] = v[i] * inv * w[lane + i * 32];

#pragma unroll
    for (int i = 0; i < 4; ++i) {
        int h = lane + i * 32;
        float rot = (i < 2) ? -y[i + 2] : y[i - 2];   // rotate_half pairing (h, h+64)
        float o = y[i] * cosT[(size_t)t * HH + h] + rot * sinT[(size_t)t * HH + h];
        out[((size_t)hd * TT + t) * HH + h] = f32_to_bf16u(o);
    }
}

// ---------------- V: f32 [T][KV*H] -> bf16 [KV][H][T] (transposed) ----------
__global__ void v_transpose_kernel(const float* __restrict__ vsrc,
                                   unsigned short* __restrict__ vt)
{
    int idx = blockIdx.x * 256 + threadIdx.x;     // T*512 total
    int t  = idx >> 9;
    int kh = idx & 511;                           // kvhead*128 + h
    vt[(size_t)kh * TT + t] = f32_to_bf16u(vsrc[idx]);
}

// ---------------- causal flash attention ----------------
// grid = (T/16, N), block = 32 (one wave). Q tile 16x128, key blocks of 32,
// online softmax, O accumulated as 8 (16x16) f32 tiles across H.
__global__ __launch_bounds__(32) void attn_kernel(
    const unsigned short* __restrict__ Qh,  // [N][T][H]
    const unsigned short* __restrict__ Kh,  // [KV][T][H]
    const unsigned short* __restrict__ Vt,  // [KV][H][T]
    unsigned short* __restrict__ Ob)        // [T][N*H] bf16
{
    __shared__ unsigned short Plb[16 * 32];   // P tile (bf16) C-layout -> A-layout
    const int lane = threadIdx.x;
    const int hl   = lane >> 4;
    const int l16  = lane & 15;
    const int q0   = blockIdx.x * 16;
    const int n    = blockIdx.y;
    const int kvh  = n / (NQ / NKV);
    const float scale = 0.08838834764831845f;   // 1/sqrt(128)

    const __bf16* Qp = (const __bf16*)Qh + (size_t)n   * TT * HH;
    const __bf16* Kp = (const __bf16*)Kh + (size_t)kvh * TT * HH;
    const __bf16* Vp = (const __bf16*)Vt + (size_t)kvh * HH * TT;

    // preload Q fragments: 4 K-chunks of 32 over H=128
    v16bf qa[4];
#pragma unroll
    for (int hc = 0; hc < 4; ++hc)
        qa[hc] = load_a_frag(Qp + (size_t)(q0 + l16) * HH + hc * 32 + hl * 16);

    v8f O[8];
    float mrow[8], lrow[8];
#pragma unroll
    for (int i = 0; i < 8; ++i) { v8f z = {}; O[i] = z; mrow[i] = -3.0e38f; lrow[i] = 0.0f; }

    for (int s0 = 0; s0 <= q0 + 15; s0 += 32) {
        // ---- scores S = Q * K^T for 32 keys (two 16x16 tiles) ----
        v8f S[2];
#pragma unroll
        for (int st = 0; st < 2; ++st) {
            v8f acc = {};
#pragma unroll
            for (int hc = 0; hc < 4; ++hc) {
                // B fragment: col = key s0+st*16+l16, contract h = hc*32+hl*16+j
                v16bf kb = *(const v16bf*)(Kp + (size_t)(s0 + st * 16 + l16) * HH
                                              + hc * 32 + hl * 16);
                acc = __builtin_amdgcn_wmma_f32_16x16x32_bf16(false, qa[hc], false, kb,
                                                              (short)0, acc, false, false);
            }
            S[st] = acc;
        }

        // ---- scale, causal mask, online softmax ----
        float p0[8], p1[8];
#pragma unroll
        for (int r = 0; r < 8; ++r) {
            const int qidx = q0 + r + hl * 8;
            float sv0 = S[0][r] * scale;
            float sv1 = S[1][r] * scale;
            if (s0 + l16      > qidx) sv0 = -3.0e38f;
            if (s0 + 16 + l16 > qidx) sv1 = -3.0e38f;
            float mx = fmaxf(sv0, sv1);
#pragma unroll
            for (int m = 8; m >= 1; m >>= 1) mx = fmaxf(mx, __shfl_xor(mx, m, 32));
            const float mnew = fmaxf(mrow[r], mx);
            const float corr = __expf(mrow[r] - mnew);
            p0[r] = __expf(sv0 - mnew);
            p1[r] = __expf(sv1 - mnew);
            float rs = p0[r] + p1[r];
#pragma unroll
            for (int m = 8; m >= 1; m >>= 1) rs += __shfl_xor(rs, m, 32);
            lrow[r] = lrow[r] * corr + rs;
            mrow[r] = mnew;
#pragma unroll
            for (int ht = 0; ht < 8; ++ht) O[ht][r] *= corr;
        }

        // ---- re-stripe P (C-layout) into A-layout via LDS, as bf16 ----
        __syncthreads();   // prior iteration's reads done
#pragma unroll
        for (int r = 0; r < 8; ++r) {
            Plb[(r + hl * 8) * 32 +      l16] = f32_to_bf16u(p0[r]);
            Plb[(r + hl * 8) * 32 + 16 + l16] = f32_to_bf16u(p1[r]);
        }
        __syncthreads();

        // A-layout read: row m = l16, one contiguous 16-element run
        v16bf pa = load_a_frag((const __bf16*)Plb + l16 * 32 + hl * 16);

        // ---- O += P * V  (8 N-tiles over H) ----
#pragma unroll
        for (int ht = 0; ht < 8; ++ht) {
            // B fragment: col h = ht*16+l16, contract t = s0+hl*16+j (contiguous in Vt)
            v16bf vb = *(const v16bf*)(Vp + (size_t)(ht * 16 + l16) * TT + s0 + hl * 16);
            O[ht] = __builtin_amdgcn_wmma_f32_16x16x32_bf16(false, pa, false, vb,
                                                            (short)0, O[ht], false, false);
        }
    }

    // ---- epilogue: normalize and store bf16 [T][N*H] ----
#pragma unroll
    for (int ht = 0; ht < 8; ++ht) {
        const int h = n * HH + ht * 16 + l16;
#pragma unroll
        for (int r = 0; r < 8; ++r) {
            float o = O[ht][r] / lrow[r];
            Ob[(size_t)(q0 + r + hl * 8) * NHD + h] = f32_to_bf16u(o);
        }
    }
}

// ---------------- host launcher ----------------
extern "C" void kernel_launch(void* const* d_in, const int* in_sizes, int n_in,
                              void* d_out, int out_size, void* d_ws, size_t ws_size,
                              hipStream_t stream) {
    (void)in_sizes; (void)n_in; (void)out_size; (void)ws_size;
    const float* hidden = (const float*)d_in[0];
    const float* cosT   = (const float*)d_in[1];
    const float* sinT   = (const float*)d_in[2];
    const float* Wq     = (const float*)d_in[3];
    const float* Wk     = (const float*)d_in[4];
    const float* Wv     = (const float*)d_in[5];
    const float* Wo     = (const float*)d_in[6];
    const float* qnw    = (const float*)d_in[7];
    const float* knw    = (const float*)d_in[8];
    float* out = (float*)d_out;

    char* ws = (char*)d_ws;
    size_t off = 0;
    auto alloc = [&](size_t bytes) -> char* {
        char* p = ws + off;
        off = (off + bytes + 255) & ~(size_t)255;
        return p;
    };
    unsigned short* hidB = (unsigned short*)alloc((size_t)TT * DD * 2);
    unsigned short* wqT  = (unsigned short*)alloc((size_t)NHD * DD * 2);  // [NHD][DD]
    unsigned short* wkT  = (unsigned short*)alloc((size_t)KHD * DD * 2);  // [KHD][DD]
    unsigned short* wvT  = (unsigned short*)alloc((size_t)KHD * DD * 2);  // [KHD][DD]
    unsigned short* woT  = (unsigned short*)alloc((size_t)DD * NHD * 2);  // [DD][NHD]
    float*          qF   = (float*)         alloc((size_t)TT * NHD * 4);
    float*          kF   = (float*)         alloc((size_t)TT * KHD * 4);
    float*          vF   = (float*)         alloc((size_t)TT * KHD * 4);
    unsigned short* qH   = (unsigned short*)alloc((size_t)NQ  * TT * HH * 2);
    unsigned short* kH   = (unsigned short*)alloc((size_t)NKV * TT * HH * 2);
    unsigned short* vT   = (unsigned short*)alloc((size_t)NKV * HH * TT * 2);
    unsigned short* aB   = (unsigned short*)alloc((size_t)TT * NHD * 2);

    // 1) conversions: hidden flat; weights converted + transposed (B^T layout)
    cvt_bf16_kernel<<<(TT * DD + 255) / 256, 256, 0, stream>>>(hidden, hidB, TT * DD);
    cvt_bf16_t_kernel<<<(DD * NHD + 255) / 256, 256, 0, stream>>>(Wq, wqT, DD, NHD);
    cvt_bf16_t_kernel<<<(DD * KHD + 255) / 256, 256, 0, stream>>>(Wk, wkT, DD, KHD);
    cvt_bf16_t_kernel<<<(DD * KHD + 255) / 256, 256, 0, stream>>>(Wv, wvT, DD, KHD);
    cvt_bf16_t_kernel<<<(NHD * DD + 255) / 256, 256, 0, stream>>>(Wo, woT, NHD, DD);

    // 2) QKV projections (bf16 WMMA, f32 accumulate), no LDS, contiguous frags
    gemm_bf16_tn<<<dim3(TT / 32, NHD / 256), 128, 0, stream>>>(hidB, wqT, qF, DD, NHD);
    gemm_bf16_tn<<<dim3(TT / 32, KHD / 256), 128, 0, stream>>>(hidB, wkT, kF, DD, KHD);
    gemm_bf16_tn<<<dim3(TT / 32, KHD / 256), 128, 0, stream>>>(hidB, wvT, vF, DD, KHD);

    // 3) RMSNorm + RoPE -> bf16 head-major layouts; V -> [KV][H][T]
    qk_post_kernel<<<dim3(TT, NQ),  32, 0, stream>>>(qF, cosT, sinT, qnw, qH, NQ);
    qk_post_kernel<<<dim3(TT, NKV), 32, 0, stream>>>(kF, cosT, sinT, knw, kH, NKV);
    v_transpose_kernel<<<(TT * KHD) / 256, 256, 0, stream>>>(vF, vT);

    // 4) causal flash attention
    attn_kernel<<<dim3(TT / 16, NQ), 32, 0, stream>>>(qH, kH, vT, aB);

    // 5) output projection -> f32 d_out
    gemm_bf16_tn<<<dim3(TT / 32, DD / 256), 128, 0, stream>>>(aB, woT, out, NHD, DD);
}
